// SelfAttention3D_15410342658217
// MI455X (gfx1250) — matrix operand order repaired and checked by reading, not compiled, not run on hardware
//
#include <hip/hip_runtime.h>

#define BB 4
#define CC 256
#define C8 32
#define NN 4096

#define LOG2E 1.44269504088896340736f
#define VPAD 40                         // LDS V row stride (elems): 80B, bank-conflict-free

typedef __attribute__((ext_vector_type(16))) __bf16 v16bf;
typedef __attribute__((ext_vector_type(8)))  __bf16 v8bf;
typedef __attribute__((ext_vector_type(8)))  float  v8f;

static __device__ __forceinline__ v16bf cat8(v8bf lo, v8bf hi) {
  return __builtin_shufflevector(lo, hi, 0,1,2,3,4,5,6,7,8,9,10,11,12,13,14,15);
}

// ---- raw VALU max (no canonicalization) ------------------------------------
static __device__ __forceinline__ float vmax2(float a, float b) {
  float r;
  asm("v_max_num_f32 %0, %1, %2" : "=v"(r) : "v"(a), "v"(b));
  return r;
}
// ---- fused DPP-swizzle + max: one instruction per butterfly step -----------
static __device__ __forceinline__ float vmax_dpp_qp1032(float x) {
  float r;
  asm("v_max_num_f32_dpp %0, %1, %2 quad_perm:[1,0,3,2] row_mask:0xf bank_mask:0xf bound_ctrl:1"
      : "=v"(r) : "v"(x), "v"(x));
  return r;
}
static __device__ __forceinline__ float vmax_dpp_qp2301(float x) {
  float r;
  asm("v_max_num_f32_dpp %0, %1, %2 quad_perm:[2,3,0,1] row_mask:0xf bank_mask:0xf bound_ctrl:1"
      : "=v"(r) : "v"(x), "v"(x));
  return r;
}
static __device__ __forceinline__ float vmax_dpp_hmirror(float x) {
  float r;
  asm("v_max_num_f32_dpp %0, %1, %2 row_half_mirror row_mask:0xf bank_mask:0xf bound_ctrl:1"
      : "=v"(r) : "v"(x), "v"(x));
  return r;
}
static __device__ __forceinline__ float vmax_dpp_mirror(float x) {
  float r;
  asm("v_max_num_f32_dpp %0, %1, %2 row_mirror row_mask:0xf bank_mask:0xf bound_ctrl:1"
      : "=v"(r) : "v"(x), "v"(x));
  return r;
}
// 16-lane row-half max: 4 fused instructions (involution pairings).
static __device__ __forceinline__ float hmax16(float x) {
  x = vmax_dpp_qp1032(x);
  x = vmax_dpp_qp2301(x);
  x = vmax_dpp_hmirror(x);
  x = vmax_dpp_mirror(x);
  return x;
}

// ---------------------------------------------------------------------------
// Kernel 1: fused 1x1x1-conv projections (fp32 math), emits bf16 Q/K/V.
//   Q,K : [B][N][32]  (position-major rows -> contiguous WMMA A/B tiles)
//   V   : [B][C][N]   (channel-major -> contiguous PV B tiles along j)
// Q is pre-scaled by log2(e) so attention uses exp2 (v_exp_f32 natively).
// ---------------------------------------------------------------------------
__global__ __launch_bounds__(256)
void sa3d_proj_kernel(const float* __restrict__ x,
                      const float* __restrict__ Wq, const float* __restrict__ bq,
                      const float* __restrict__ Wk, const float* __restrict__ bk,
                      const float* __restrict__ Wv, const float* __restrict__ bv,
                      __bf16* __restrict__ Qb, __bf16* __restrict__ Kb,
                      __bf16* __restrict__ Vb)
{
  __shared__ float xs[CC * 16];
  const int t  = threadIdx.x;
  const int b  = blockIdx.x >> 8;
  const int n0 = (blockIdx.x & 255) << 4;

  const float* xb = x + (size_t)b * CC * NN + n0;
#pragma unroll
  for (int i = 0; i < 16; ++i) {
    int idx = i * 256 + t;
    int c = idx >> 4, n = idx & 15;
    xs[idx] = xb[(size_t)c * NN + n];
  }
  __syncthreads();

  const int n  = t & 15;
  const int og = t >> 4;              // 16 groups x 20 outputs = 320 channels
  for (int kk = 0; kk < 20; ++kk) {
    const int o = og * 20 + kk;       // 0..319: [0,32)=Q [32,64)=K [64,320)=V
    const float* wrow;
    float acc;
    if (o < 32)      { wrow = Wq + o * CC;        acc = bq[o]; }
    else if (o < 64) { wrow = Wk + (o - 32) * CC; acc = bk[o - 32]; }
    else             { wrow = Wv + (o - 64) * CC; acc = bv[o - 64]; }
#pragma unroll 4
    for (int c0 = 0; c0 < CC; c0 += 8) {
      v8f w8 = *(const v8f*)(wrow + c0);
#pragma unroll
      for (int u = 0; u < 8; ++u) acc += w8[u] * xs[(c0 + u) * 16 + n];
    }
    if (o < 32)
      Qb[(size_t)(b * NN + n0 + n) * C8 + o] = (__bf16)(acc * LOG2E);
    else if (o < 64)
      Kb[(size_t)(b * NN + n0 + n) * C8 + (o - 32)] = (__bf16)acc;
    else
      Vb[(size_t)(b * CC + (o - 64)) * NN + n0 + n] = (__bf16)acc;
  }
}

// ---------------------------------------------------------------------------
// Kernel 2: two-pass flash attention + gamma*out + x residual.
// 8 waves / block = 8 consecutive 16-query tiles of one batch, sharing the
// V stream through double-buffered LDS (8x less L2 traffic on the PV operand).
//   Pass 1 (per wave): 64 keys/chunk: 4 S-wmma -> fused DPP butterfly max.
//   Pass 2 (block-synchronous over j): stage V[0..256)[j..j+32) to LDS
//   cooperatively, P' = exp2(S - m) -> bf16, O' += P' x V (16 wmma from LDS),
//   l += P' x ones (1 wmma). Epilogue: 8 reciprocals, gamma, residual.
// ---------------------------------------------------------------------------
__global__ __launch_bounds__(256)
void sa3d_attn_kernel(const __bf16* __restrict__ Qb,
                      const __bf16* __restrict__ Kb,
                      const __bf16* __restrict__ Vb,
                      const float*  __restrict__ x,
                      const float*  __restrict__ gamma,
                      float* __restrict__ out)
{
  __shared__ __align__(16) __bf16 vbuf[2 * CC * VPAD];   // 40KB V ping-pong
  __shared__ __align__(16) __bf16 pbuf[8 * 16 * 32];     //  8KB P slices

  const int t    = threadIdx.x;
  const int lane = t & 31;
  const int wave = t >> 5;
  const int tile = blockIdx.x * 8 + wave;       // 1024 query tiles total
  const int b    = tile >> 8;
  const int i0   = (tile & 255) << 4;
  const int col  = lane & 15;
  const int h    = lane >> 4;                   // lane half (C-layout M offset)

  // Resident Q A-tile (16 queries x 32 ch).
  const __bf16* qrow = Qb + (size_t)(b * NN + i0 + col) * C8;
  const v16bf qa = cat8(*(const v8bf*)(qrow + h * 16),
                        *(const v8bf*)(qrow + h * 16 + 8));

  v8f zc;
#pragma unroll
  for (int v = 0; v < 8; ++v) zc[v] = 0.0f;

  v16bf onesB;                                  // 32x16 all-ones B tile
#pragma unroll
  for (int v = 0; v < 16; ++v) onesB[v] = (__bf16)1.0f;

  const __bf16* kb_base = Kb + (size_t)b * NN * C8;
  const __bf16* vb_base = Vb + (size_t)b * CC * NN;
  __bf16* pw = pbuf + wave * (16 * 32);

  // -------- Pass 1: row maxima, 64 keys per chunk (per-wave, no sync) ------
  float m[8];
#pragma unroll
  for (int v = 0; v < 8; ++v) m[v] = -1e30f;

  for (int j = 0; j < NN; j += 64) {
    const __bf16* kp = kb_base + (size_t)(j + col) * C8 + h * 16;
    v8f s[4];
#pragma unroll
    for (int u = 0; u < 4; ++u) {
      const __bf16* k = kp + (size_t)(16 * u) * C8;
      const v16bf kT = cat8(*(const v8bf*)k, *(const v8bf*)(k + 8));
      s[u] = __builtin_amdgcn_wmma_f32_16x16x32_bf16(false, qa, false, kT,
                                                     (short)0, zc, false, false);
    }
#pragma unroll
    for (int v = 0; v < 8; ++v) {
      const float tmx = vmax2(vmax2(s[0][v], s[1][v]), vmax2(s[2][v], s[3][v]));
      m[v] = vmax2(m[v], hmax16(tmx));
    }
  }

  // -------- Pass 2: O' = P' x V and l = P' x 1, V staged through LDS -------
  v8f o[16];
#pragma unroll
  for (int tt = 0; tt < 16; ++tt)
#pragma unroll
    for (int v = 0; v < 8; ++v) o[tt][v] = 0.0f;
  v8f lacc = zc;

  int jt = 0;
  for (int j = 0; j < NN; j += 32, ++jt) {
    // Cooperative V staging: thread t copies channel-row t (64B) of chunk j.
    {
      const __bf16* vsrc = vb_base + (size_t)t * NN + j;
      __bf16* vdst = vbuf + (jt & 1) * (CC * VPAD) + t * VPAD;
      *(v8bf*)(vdst)      = *(const v8bf*)(vsrc);
      *(v8bf*)(vdst + 8)  = *(const v8bf*)(vsrc + 8);
      *(v8bf*)(vdst + 16) = *(const v8bf*)(vsrc + 16);
      *(v8bf*)(vdst + 24) = *(const v8bf*)(vsrc + 24);
    }

    // Scores for this wave's 16 queries (overlaps the staging copies).
    const __bf16* k0 = kb_base + (size_t)(j + col) * C8 + h * 16;
    const __bf16* k1 = k0 + 16 * C8;
    const v16bf kB0 = cat8(*(const v8bf*)k0, *(const v8bf*)(k0 + 8));
    const v16bf kB1 = cat8(*(const v8bf*)k1, *(const v8bf*)(k1 + 8));

    v8f s0 = __builtin_amdgcn_wmma_f32_16x16x32_bf16(false, qa, false, kB0,
                                                     (short)0, zc, false, false);
    v8f s1 = __builtin_amdgcn_wmma_f32_16x16x32_bf16(false, qa, false, kB1,
                                                     (short)0, zc, false, false);

    // Unnormalized P' = exp2(S - m): C-layout -> row-major bf16 16x32 in LDS.
#pragma unroll
    for (int v = 0; v < 8; ++v) {
      const int r = v + 8 * h;
      pw[r * 32 + col]      = (__bf16)__builtin_amdgcn_exp2f(s0[v] - m[v]);
      pw[r * 32 + 16 + col] = (__bf16)__builtin_amdgcn_exp2f(s1[v] - m[v]);
    }
    asm volatile("" ::: "memory");   // same-wave DS ops are in-order; stop reordering
    const __bf16* pr = pw + col * 32 + h * 16;
    const v16bf pa = cat8(*(const v8bf*)pr, *(const v8bf*)(pr + 8));

    // Softmax denominator via matrix pipe: l += P' x ones.
    lacc = __builtin_amdgcn_wmma_f32_16x16x32_bf16(false, pa, false, onesB,
                                                   (short)0, lacc, false, false);

    // Publish staged V (barrier waits each wave's own DS/load counters first;
    // double buffering makes one barrier per chunk sufficient).
    __syncthreads();

    // O' += P'(16x32) x V(32x16 per channel tile) from padded LDS rows.
    const __bf16* vls = vbuf + (jt & 1) * (CC * VPAD);
#pragma unroll
    for (int tt = 0; tt < 16; ++tt) {
      const __bf16* vp = vls + (tt * 16 + col) * VPAD + h * 16;
      const v16bf vB = cat8(*(const v8bf*)vp, *(const v8bf*)(vp + 8));
      o[tt] = __builtin_amdgcn_wmma_f32_16x16x32_bf16(false, pa, false, vB,
                                                      (short)0, o[tt], false, false);
    }
  }

  // Epilogue: normalize (l C-tile rows == O rows), gamma, residual.
  const float g = gamma[0];
  float rl[8];
#pragma unroll
  for (int v = 0; v < 8; ++v) rl[v] = 1.0f / lacc[v];

#pragma unroll
  for (int tt = 0; tt < 16; ++tt) {
    const int c = tt * 16 + col;
    const size_t base = (size_t)(b * CC + c) * NN + i0 + h * 8;
    const v8f xr = *(const v8f*)(x + base);
    v8f r;
#pragma unroll
    for (int v = 0; v < 8; ++v) r[v] = g * (o[tt][v] * rl[v]) + xr[v];
    *(v8f*)(out + base) = r;
  }
}

// ---------------------------------------------------------------------------
extern "C" void kernel_launch(void* const* d_in, const int* in_sizes, int n_in,
                              void* d_out, int out_size, void* d_ws, size_t ws_size,
                              hipStream_t stream) {
  const float* x     = (const float*)d_in[0];
  const float* Wq    = (const float*)d_in[1];
  const float* bq    = (const float*)d_in[2];
  const float* Wk    = (const float*)d_in[3];
  const float* bk    = (const float*)d_in[4];
  const float* Wv    = (const float*)d_in[5];
  const float* bv    = (const float*)d_in[6];
  const float* gamma = (const float*)d_in[7];
  float* out = (float*)d_out;

  __bf16* ws = (__bf16*)d_ws;                 // 10 MB used
  __bf16* Qb = ws;                            // [B][N][32]
  __bf16* Kb = ws + (size_t)BB * NN * C8;     // [B][N][32]
  __bf16* Vb = ws + (size_t)2 * BB * NN * C8; // [B][C][N]

  sa3d_proj_kernel<<<BB * (NN / 16), 256, 0, stream>>>(
      x, Wq, bq, Wk, bk, Wv, bv, Qb, Kb, Vb);

  sa3d_attn_kernel<<<(BB * (NN / 16)) / 8, 256, 0, stream>>>(
      Qb, Kb, Vb, x, gamma, out);
}
